// ColorNetwork_9096740733262
// MI455X (gfx1250) — compile-verified, matrix operand order checked by reading, not run on hardware
//
#include <hip/hip_runtime.h>
#include <math.h>

// ---------------------------------------------------------------------------
// CDNA5 (gfx1250) WMMA types & helpers
// ---------------------------------------------------------------------------
typedef __attribute__((ext_vector_type(16))) __bf16 v16bf;
typedef __attribute__((ext_vector_type(8)))  __bf16 v8bf;
typedef __attribute__((ext_vector_type(4)))  __bf16 v4bf;
typedef __attribute__((ext_vector_type(8)))  float  v8f;

__device__ __forceinline__ __bf16 f2bf(float f) {
    // round-to-nearest-even fp32 -> bf16 (finite inputs)
    unsigned u = __builtin_bit_cast(unsigned, f);
    u += 0x7FFFu + ((u >> 16) & 1u);
    unsigned short h = (unsigned short)(u >> 16);
    return __builtin_bit_cast(__bf16, h);
}
__device__ __forceinline__ float bf2f(__bf16 h) {
    unsigned short s = __builtin_bit_cast(unsigned short, h);
    unsigned u = ((unsigned)s) << 16;
    return __builtin_bit_cast(float, u);
}

// A-fragment for V_WMMA_F32_16X16X32_BF16, source already bf16 row-major.
// Lane L holds row M = L&15. For K-block starting at k0:
//   lanes 0-15 : elems 0-7 -> k0+0..7,  elems 8-15 -> k0+16..23
//   lanes 16-31: elems 0-7 -> k0+8..15, elems 8-15 -> k0+24..31
// => two aligned 16-byte loads per lane.
__device__ __forceinline__ v16bf a_frag_bf(const __bf16* Arow, int k0, int lane) {
    const int kb = k0 + ((lane & 16) ? 8 : 0);
    v8bf lo = *(const v8bf*)(Arow + kb);
    v8bf hi = *(const v8bf*)(Arow + kb + 16);
    return __builtin_shufflevector(lo, hi, 0, 1, 2, 3, 4, 5, 6, 7,
                                   8, 9, 10, 11, 12, 13, 14, 15);
}

// A-fragment gathered from f32 row-major (only used by the tiny 64x64 GEMMs).
__device__ __forceinline__ v16bf a_frag_f32(const float* Arow, int k0, int lane) {
    v16bf f;
    const int kb = k0 + ((lane & 16) ? 8 : 0);
#pragma unroll
    for (int i = 0; i < 8; ++i)  f[i] = f2bf(Arow[kb + i]);
#pragma unroll
    for (int i = 8; i < 16; ++i) f[i] = f2bf(Arow[kb + 8 + i]);   // kb+16+(i-8)
    return f;
}

// B-fragment from pre-packed weights: layout [kb][nt][lane][16] bf16
// (one aligned 32-byte load per lane, zero conversion in the hot loop).
__device__ __forceinline__ v16bf b_frag_pk(const __bf16* Wpk, int nNT,
                                           int kb, int nt, int lane) {
    return *(const v16bf*)(Wpk + (((size_t)(kb * nNT + nt) * 32 + lane) << 4));
}

__device__ __forceinline__ v8f wmma_bf16(v16bf a, v16bf b, v8f c) {
    return __builtin_amdgcn_wmma_f32_16x16x32_bf16(
        /*neg_a=*/false, a, /*neg_b=*/false, b,
        /*c_mod=*/(short)0, c, /*reuse_a=*/false, /*reuse_b=*/false);
}

// C/D layout: lane L -> column (L&15); element r -> row r + ((L&16)?8:0).

// ---------------------------------------------------------------------------
// Prep kernels (run once per launch; trivially cheap)
// ---------------------------------------------------------------------------

// Row-major f32 W[K,N] -> fragment-packed bf16 [kb][nt][lane][16].
__global__ void pack_w_kernel(const float* __restrict__ W, int K, int N,
                              __bf16* __restrict__ out) {
    const int total = K * N;
    const int idx = blockIdx.x * blockDim.x + threadIdx.x;
    if (idx >= total) return;
    const int nNT = N >> 4;
    const int i = idx & 15;
    const int lane = (idx >> 4) & 31;
    const int rest = idx >> 9;
    const int nt = rest % nNT;
    const int kb = rest / nNT;
    const int k = kb * 32 + ((lane & 16) ? 8 : 0) + ((i < 8) ? i : (8 + i));
    const int n = nt * 16 + (lane & 15);
    out[idx] = f2bf(W[(size_t)k * N + n]);
}

__global__ void cvt_f32_to_bf16_kernel(const float* __restrict__ in,
                                       __bf16* __restrict__ out, long n) {
    long i = (long)blockIdx.x * blockDim.x + threadIdx.x;
    if (i < n) out[i] = f2bf(in[i]);
}

__global__ void zero_kernel(float* __restrict__ a, long na,
                            float* __restrict__ b, long nb) {
    long i = (long)blockIdx.x * blockDim.x + threadIdx.x;
    if (i < na) a[i] = 0.0f;
    if (i < nb) b[i] = 0.0f;
}

// ---------------------------------------------------------------------------
// Edge aggregation: agg_v[dst] += mv[src]; thread = (edge, 4-elem chunk).
// mv is bf16 (halves gather traffic); accumulation in f32 atomics (L2-resident).
// ---------------------------------------------------------------------------
__global__ void edge_agg_kernel(const __bf16* __restrict__ mv,
                                const int* __restrict__ esrc,
                                const int* __restrict__ edst,
                                float* __restrict__ agg_v, long total) {
    long idx = (long)blockIdx.x * blockDim.x + threadIdx.x;
    if (idx >= total) return;
    const int e = (int)(idx >> 4);
    const int c = (int)(idx & 15);
    const int s = esrc[e];
    const int d = edst[e];
    const v4bf v = *(const v4bf*)(mv + (size_t)s * 64 + c * 4);
    float* o = agg_v + (size_t)d * 64 + c * 4;
    atomicAdd(o + 0, bf2f(v[0]));
    atomicAdd(o + 1, bf2f(v[1]));
    atomicAdd(o + 2, bf2f(v[2]));
    atomicAdd(o + 3, bf2f(v[3]));
}

// ---------------------------------------------------------------------------
// out[64,64] = relu(A[64,64] @ W + bias), W pre-packed bf16. 1 block, 4 waves.
// ---------------------------------------------------------------------------
__global__ void __launch_bounds__(128) mm64_relu_kernel(
    const float* __restrict__ A, const __bf16* __restrict__ Wpk,
    const float* __restrict__ bias, float* __restrict__ out) {
    const int wave = threadIdx.x >> 5, lane = threadIdx.x & 31;
    const float* Arow = A + (size_t)(wave * 16 + (lane & 15)) * 64;
    v8f acc[4] = {};
#pragma unroll
    for (int kb = 0; kb < 2; ++kb) {
        v16bf a = a_frag_f32(Arow, kb * 32, lane);
#pragma unroll
        for (int nt = 0; nt < 4; ++nt)
            acc[nt] = wmma_bf16(a, b_frag_pk(Wpk, 4, kb, nt, lane), acc[nt]);
    }
    const int colb = lane & 15;
    const int rowoff = (lane & 16) ? 8 : 0;
    const int rowbase = wave * 16;
#pragma unroll
    for (int nt = 0; nt < 4; ++nt) {
        const float bv = bias[nt * 16 + colb];
#pragma unroll
        for (int r = 0; r < 8; ++r)
            out[(size_t)(rowbase + r + rowoff) * 64 + nt * 16 + colb] =
                fmaxf(acc[nt][r] + bv, 0.0f);
    }
}

// ---------------------------------------------------------------------------
// Fused vertex step, one 16-vertex tile per wave:
//   mv_new = relu([agg_v ; c_msg[color]] @ W_vu + b_vu)   (K=128, 16 WMMA)
//   v_msg  = relu(mv_new @ W_vm + b_vm)                   (K=64,   8 WMMA)
//   agg_c[color[v]] += v_msg[v]
// ---------------------------------------------------------------------------
__global__ void __launch_bounds__(128) vertex_update_kernel(
    const float* __restrict__ agg_v, const float* __restrict__ c_msg,
    const int* __restrict__ color,
    const __bf16* __restrict__ pW_vu, const float* __restrict__ b_vu,
    const __bf16* __restrict__ pW_vm, const float* __restrict__ b_vm,
    __bf16* __restrict__ mv_out, float* __restrict__ agg_c, int n_tiles) {
    __shared__ __bf16 ldsA[4][16 * 128];   // 16 KB: concat input tile (bf16)
    __shared__ __bf16 ldsH[4][16 * 64];    //  8 KB: mv_new tile (bf16)

    const int wave = threadIdx.x >> 5, lane = threadIdx.x & 31;
    const int tile = blockIdx.x * 4 + wave;
    const bool active = (tile < n_tiles);
    const int vbase = active ? tile * 16 : 0;

    // Stage A = [agg_v | c_msg[color]] as bf16 (convert during coalesced pass)
    __bf16* A = ldsA[wave];
    for (int i = lane; i < 512; i += 32) {
        const int row = i >> 5, c4 = i & 31;
        const int v = vbase + row;
        float4 val;
        if (c4 < 16) val = *(const float4*)(agg_v + (size_t)v * 64 + c4 * 4);
        else         val = *(const float4*)(c_msg + (size_t)color[v] * 64 + (c4 - 16) * 4);
        v4bf pk;
        pk[0] = f2bf(val.x); pk[1] = f2bf(val.y);
        pk[2] = f2bf(val.z); pk[3] = f2bf(val.w);
        *(v4bf*)(A + row * 128 + c4 * 4) = pk;
    }
    __syncthreads();

    // GEMM 1: (16x128) @ (128x64)
    const __bf16* Arow = A + (lane & 15) * 128;
    v8f acc[4] = {};
#pragma unroll
    for (int kb = 0; kb < 4; ++kb) {
        v16bf a = a_frag_bf(Arow, kb * 32, lane);
#pragma unroll
        for (int nt = 0; nt < 4; ++nt)
            acc[nt] = wmma_bf16(a, b_frag_pk(pW_vu, 4, kb, nt, lane), acc[nt]);
    }

    const int colb = lane & 15;
    const int rowoff = (lane & 16) ? 8 : 0;
    __bf16* H = ldsH[wave];
#pragma unroll
    for (int nt = 0; nt < 4; ++nt) {
        const float bv = b_vu[nt * 16 + colb];
#pragma unroll
        for (int r = 0; r < 8; ++r)
            H[(r + rowoff) * 64 + nt * 16 + colb] =
                f2bf(fmaxf(acc[nt][r] + bv, 0.0f));
    }
    __syncthreads();

    // Coalesced store of mv_new tile (contiguous 2 KB per tile)
    if (active) {
        const uint2* src = (const uint2*)H;                 // 256 x 8B
        uint2* dst = (uint2*)(mv_out + (size_t)vbase * 64);
        for (int i = lane; i < 256; i += 32) dst[i] = src[i];
    }

    // GEMM 2: (16x64) @ (64x64) from LDS-resident bf16 mv_new
    const __bf16* Hrow = H + (lane & 15) * 64;
    v8f acc2[4] = {};
#pragma unroll
    for (int kb = 0; kb < 2; ++kb) {
        v16bf a = a_frag_bf(Hrow, kb * 32, lane);
#pragma unroll
        for (int nt = 0; nt < 4; ++nt)
            acc2[nt] = wmma_bf16(a, b_frag_pk(pW_vm, 4, kb, nt, lane), acc2[nt]);
    }

    if (active) {
#pragma unroll
        for (int r = 0; r < 8; ++r) {
            const int row = r + rowoff;
            const int cidx = color[vbase + row];
#pragma unroll
            for (int nt = 0; nt < 4; ++nt) {
                float v = fmaxf(acc2[nt][r] + b_vm[nt * 16 + colb], 0.0f);
                atomicAdd(&agg_c[(size_t)cidx * 64 + nt * 16 + colb], v);
            }
        }
    }
}

// ---------------------------------------------------------------------------
// Voting MLP: h1=relu(mv@W1+b1); h2=relu(h1@W2+b2); vote=h2@W3; sum-reduce.
// ---------------------------------------------------------------------------
__global__ void __launch_bounds__(128) vote_kernel(
    const __bf16* __restrict__ mv,
    const __bf16* __restrict__ pW1, const float* __restrict__ b1,
    const __bf16* __restrict__ pW2, const float* __restrict__ b2,
    const float* __restrict__ W3, float* __restrict__ vote_sum, int n_tiles) {
    __shared__ __bf16 ldsH[4][16 * 64];
    __shared__ __bf16 ldsG[4][16 * 64];

    const int wave = threadIdx.x >> 5, lane = threadIdx.x & 31;
    const int tile = blockIdx.x * 4 + wave;
    const bool active = (tile < n_tiles);
    const int vbase = active ? tile * 16 : 0;
    const int colb = lane & 15;
    const int rowoff = (lane & 16) ? 8 : 0;

    // Layer 1: A rows straight from global bf16 mv (L2-resident, aligned b128)
    const __bf16* Arow = mv + (size_t)(vbase + (lane & 15)) * 64;
    v8f acc[4] = {};
#pragma unroll
    for (int kb = 0; kb < 2; ++kb) {
        v16bf a = a_frag_bf(Arow, kb * 32, lane);
#pragma unroll
        for (int nt = 0; nt < 4; ++nt)
            acc[nt] = wmma_bf16(a, b_frag_pk(pW1, 4, kb, nt, lane), acc[nt]);
    }
    __bf16* H = ldsH[wave];
#pragma unroll
    for (int nt = 0; nt < 4; ++nt) {
        const float bv = b1[nt * 16 + colb];
#pragma unroll
        for (int r = 0; r < 8; ++r)
            H[(r + rowoff) * 64 + nt * 16 + colb] =
                f2bf(fmaxf(acc[nt][r] + bv, 0.0f));
    }
    __syncthreads();

    // Layer 2
    const __bf16* Hrow = H + (lane & 15) * 64;
    v8f acc2[4] = {};
#pragma unroll
    for (int kb = 0; kb < 2; ++kb) {
        v16bf a = a_frag_bf(Hrow, kb * 32, lane);
#pragma unroll
        for (int nt = 0; nt < 4; ++nt)
            acc2[nt] = wmma_bf16(a, b_frag_pk(pW2, 4, kb, nt, lane), acc2[nt]);
    }
    __bf16* G = ldsG[wave];
#pragma unroll
    for (int nt = 0; nt < 4; ++nt) {
        const float bv = b2[nt * 16 + colb];
#pragma unroll
        for (int r = 0; r < 8; ++r)
            G[(r + rowoff) * 64 + nt * 16 + colb] =
                f2bf(fmaxf(acc2[nt][r] + bv, 0.0f));
    }
    __syncthreads();

    // Output head: 64-wide dot per vertex row (lanes 0..15 take one row each)
    if (active && lane < 16) {
        const __bf16* h = G + lane * 64;
        float dot = 0.0f;
#pragma unroll
        for (int d = 0; d < 64; ++d) dot += bf2f(h[d]) * W3[d];
        atomicAdd(vote_sum, dot);
    }
}

__global__ void finalize_kernel(const float* __restrict__ vote_sum,
                                const float* __restrict__ b3,
                                float* __restrict__ out, int N) {
    if (threadIdx.x == 0) {
        const float m = vote_sum[0] / (float)N + b3[0];
        out[0] = 1.0f / (1.0f + expf(-m));
    }
}

// ---------------------------------------------------------------------------
// Host orchestration (graph-capture safe: kernels + d2d memcpyAsync only)
// ---------------------------------------------------------------------------
extern "C" void kernel_launch(void* const* d_in, const int* in_sizes, int n_in,
                              void* d_out, int out_size, void* d_ws, size_t ws_size,
                              hipStream_t stream) {
    (void)n_in; (void)out_size; (void)ws_size;
    const float* M_v      = (const float*)d_in[0];
    const float* M_c      = (const float*)d_in[1];
    const int*   edge_src = (const int*)d_in[2];
    const int*   edge_dst = (const int*)d_in[3];
    const int*   color    = (const int*)d_in[4];
    const float* W_cm = (const float*)d_in[5];
    const float* b_cm = (const float*)d_in[6];
    const float* W_vm = (const float*)d_in[7];
    const float* b_vm = (const float*)d_in[8];
    const float* W_vu = (const float*)d_in[9];
    const float* b_vu = (const float*)d_in[10];
    const float* W_cu = (const float*)d_in[11];
    const float* b_cu = (const float*)d_in[12];
    const float* W1 = (const float*)d_in[13];
    const float* b1 = (const float*)d_in[14];
    const float* W2 = (const float*)d_in[15];
    const float* b2 = (const float*)d_in[16];
    const float* W3 = (const float*)d_in[17];
    const float* b3 = (const float*)d_in[18];

    const int N = in_sizes[0] / 64;
    const int E = in_sizes[2];
    const int n_tiles = (N + 15) / 16;
    const int T = 32;

    // Workspace layout (256B-aligned slots), ~26 MB total
    char* p = (char*)d_ws;
    auto alloc = [&](size_t bytes) -> void* {
        void* r = (void*)p;
        p += (bytes + 255) & ~(size_t)255;
        return r;
    };
    __bf16* mvA      = (__bf16*)alloc((size_t)N * 64 * sizeof(__bf16));
    __bf16* mvB      = (__bf16*)alloc((size_t)N * 64 * sizeof(__bf16));
    float*  agg_v    = (float*)alloc((size_t)N * 64 * sizeof(float));
    float*  mc       = (float*)alloc(64 * 64 * sizeof(float));
    float*  c_msg    = (float*)alloc(64 * 64 * sizeof(float));
    float*  agg_c    = (float*)alloc(64 * 64 * sizeof(float));
    float*  vote_sum = (float*)alloc(64);
    __bf16* pW_cm = (__bf16*)alloc(64 * 64 * sizeof(__bf16));
    __bf16* pW_vm = (__bf16*)alloc(64 * 64 * sizeof(__bf16));
    __bf16* pW_vu = (__bf16*)alloc(128 * 64 * sizeof(__bf16));
    __bf16* pW_cu = (__bf16*)alloc(64 * 64 * sizeof(__bf16));
    __bf16* pW1   = (__bf16*)alloc(64 * 64 * sizeof(__bf16));
    __bf16* pW2   = (__bf16*)alloc(64 * 64 * sizeof(__bf16));

    // One-time prep: mv -> bf16, mc copy, weights -> fragment-packed bf16
    const long nvfloat = (long)N * 64;
    cvt_f32_to_bf16_kernel<<<(int)((nvfloat + 255) / 256), 256, 0, stream>>>(
        M_v, mvA, nvfloat);
    hipMemcpyAsync(mc, M_c, (size_t)64 * 64 * sizeof(float),
                   hipMemcpyDeviceToDevice, stream);
    pack_w_kernel<<<16, 256, 0, stream>>>(W_cm, 64, 64, pW_cm);
    pack_w_kernel<<<16, 256, 0, stream>>>(W_vm, 64, 64, pW_vm);
    pack_w_kernel<<<32, 256, 0, stream>>>(W_vu, 128, 64, pW_vu);
    pack_w_kernel<<<16, 256, 0, stream>>>(W_cu, 64, 64, pW_cu);
    pack_w_kernel<<<16, 256, 0, stream>>>(W1, 64, 64, pW1);
    pack_w_kernel<<<16, 256, 0, stream>>>(W2, 64, 64, pW2);

    const int  zeroBlocks = (int)((nvfloat + 255) / 256);
    const long etotal = (long)E * 16;
    const int  edgeBlocks = (int)((etotal + 255) / 256);
    const int  vtxBlocks = (n_tiles + 3) / 4;

    __bf16* mv_cur = mvA;
    __bf16* mv_nxt = mvB;
    for (int t = 0; t < T; ++t) {
        zero_kernel<<<zeroBlocks, 256, 0, stream>>>(agg_v, nvfloat, agg_c, (long)64 * 64);
        edge_agg_kernel<<<edgeBlocks, 256, 0, stream>>>(mv_cur, edge_src, edge_dst,
                                                        agg_v, etotal);
        mm64_relu_kernel<<<1, 128, 0, stream>>>(mc, pW_cm, b_cm, c_msg);
        vertex_update_kernel<<<vtxBlocks, 128, 0, stream>>>(
            agg_v, c_msg, color, pW_vu, b_vu, pW_vm, b_vm, mv_nxt, agg_c, n_tiles);
        mm64_relu_kernel<<<1, 128, 0, stream>>>(agg_c, pW_cu, b_cu, mc);
        __bf16* tmp = mv_cur; mv_cur = mv_nxt; mv_nxt = tmp;
    }

    zero_kernel<<<1, 256, 0, stream>>>(vote_sum, 1, vote_sum, 0);
    vote_kernel<<<vtxBlocks, 128, 0, stream>>>(mv_cur, pW1, b1, pW2, b2, W3,
                                               vote_sum, n_tiles);
    finalize_kernel<<<1, 32, 0, stream>>>(vote_sum, b3, (float*)d_out, N);
}
